// Schizo_11106785427704
// MI455X (gfx1250) — matrix operand order repaired and checked by reading, not compiled, not run on hardware
//
#include <hip/hip_runtime.h>

// ---------------------------------------------------------------------------
// relu(x @ (kernel*window) + bias) as a bf16x3-split WMMA GEMM for gfx1250.
//
// Preferred path (needs 192MB workspace):
//   prep1: x -> Xhi/Xlo            (bf16, [M][K], row-major)
//   prep2: kernel*window -> Bhi/Blo (bf16, [N][K], TRANSPOSED)
//   main:  tiles staged to LDS via GLOBAL_LOAD_ASYNC_TO_LDS_B128 (ASYNCcnt),
//          fragments via ds_load_b128 in native WMMA layout, 24 wmma/K-step.
// Fallback path (small ws): fused convert-in-kernel version (round-2 kernel).
// ---------------------------------------------------------------------------

typedef __attribute__((ext_vector_type(16))) __bf16 v16bf;
typedef __attribute__((ext_vector_type(8)))  __bf16 v8bf;
typedef __attribute__((ext_vector_type(4)))  __bf16 v4bf;
typedef __attribute__((ext_vector_type(8)))  float  v8f;
typedef __attribute__((ext_vector_type(4)))  int    v4i;

// Pointer-to-int4 in explicit address spaces, matching the async builtin's
// parameter types (param0 is 'int4 __device__*' per the clang diagnostic).
typedef __attribute__((address_space(1))) v4i* gptr128;
typedef __attribute__((address_space(3))) v4i* lptr128;

#define BM 128
#define BN 128
#define BK 32
#define LDBS 40   // bf16 per LDS row: 32 data + 8 pad (80B, 16B-aligned rows)

#if __has_builtin(__builtin_amdgcn_global_load_async_to_lds_b128) && \
    __has_builtin(__builtin_amdgcn_s_wait_asynccnt)
#define HAVE_ASYNC_LDS 1
#endif

__device__ __forceinline__ __bf16 lo_part(float f, __bf16 h) {
    return (__bf16)(f - (float)h);
}

__device__ __forceinline__ void copy16_g2l(const __bf16* g, __bf16* l) {
#ifdef HAVE_ASYNC_LDS
    gptr128 gp = (gptr128)const_cast<__bf16*>(g);
    lptr128 lp = (lptr128)l;
    __builtin_amdgcn_global_load_async_to_lds_b128(gp, lp, 0, 0);
#else
    *(v8bf*)l = *(const v8bf*)g;
#endif
}

__device__ __forceinline__ void staging_fence() {
#ifdef HAVE_ASYNC_LDS
    __builtin_amdgcn_s_wait_asynccnt(0);
#endif
    __syncthreads();
}

// ---------------------------------------------------------------------------
// prep1: elementwise split x (fp32) -> Xhi/Xlo (bf16), same layout.
// ---------------------------------------------------------------------------
__global__ __launch_bounds__(256)
void prep_split_x(const float* __restrict__ X,
                  __bf16* __restrict__ Xhi, __bf16* __restrict__ Xlo)
{
    size_t i4 = (size_t)blockIdx.x * 256 + threadIdx.x;   // float4 index
    const float4 v = *(const float4*)(X + i4 * 4);
    v4bf h, l;
    h[0] = (__bf16)v.x;  l[0] = lo_part(v.x, h[0]);
    h[1] = (__bf16)v.y;  l[1] = lo_part(v.y, h[1]);
    h[2] = (__bf16)v.z;  l[2] = lo_part(v.z, h[2]);
    h[3] = (__bf16)v.w;  l[3] = lo_part(v.w, h[3]);
    *(v4bf*)(Xhi + i4 * 4) = h;
    *(v4bf*)(Xlo + i4 * 4) = l;
}

// ---------------------------------------------------------------------------
// prep2: fuse kernel*window, split, store TRANSPOSED [N][K] as bf16 hi/lo.
// Block handles a 32(k) x 128(n) tile; thread owns a 4k x 4n micro-tile.
// ---------------------------------------------------------------------------
__global__ __launch_bounds__(256)
void prep_fuse_b(const float* __restrict__ Kw, const float* __restrict__ Wn,
                 __bf16* __restrict__ BhiT, __bf16* __restrict__ BloT,
                 int N, int Kd)
{
    const int tid = threadIdx.x;
    const int gk0 = blockIdx.y * 32;
    const int gn0 = blockIdx.x * 128;
    const int rb  = (tid >> 5) * 4;   // k row base within tile
    const int c4  = tid & 31;         // float4 n-column within tile

    float pr[4][4];
#pragma unroll
    for (int rr = 0; rr < 4; ++rr) {
        size_t g = (size_t)(gk0 + rb + rr) * N + gn0 + c4 * 4;
        float4 kv = *(const float4*)(Kw + g);
        float4 wv = *(const float4*)(Wn + g);
        pr[rr][0] = kv.x * wv.x;
        pr[rr][1] = kv.y * wv.y;
        pr[rr][2] = kv.z * wv.z;
        pr[rr][3] = kv.w * wv.w;
    }
#pragma unroll
    for (int e = 0; e < 4; ++e) {
        int n = gn0 + c4 * 4 + e;
        v4bf h, l;
#pragma unroll
        for (int rr = 0; rr < 4; ++rr) {
            h[rr] = (__bf16)pr[rr][e];
            l[rr] = lo_part(pr[rr][e], h[rr]);
        }
        *(v4bf*)(BhiT + (size_t)n * Kd + gk0 + rb) = h;
        *(v4bf*)(BloT + (size_t)n * Kd + gk0 + rb) = l;
    }
}

// ---------------------------------------------------------------------------
// Main GEMM on pre-split bf16 operands: staging is pure async row copies.
// ---------------------------------------------------------------------------
__global__ __launch_bounds__(256)
void schizo_wmma_gemm_pre(const __bf16* __restrict__ Xhi,
                          const __bf16* __restrict__ Xlo,
                          const __bf16* __restrict__ BhiT,
                          const __bf16* __restrict__ BloT,
                          const float* __restrict__ Bias,
                          float* __restrict__ Out,
                          int M, int N, int Kd)
{
    __shared__ __bf16 sAhi[BM * LDBS];   // [m][k]
    __shared__ __bf16 sAlo[BM * LDBS];
    __shared__ __bf16 sBhi[BN * LDBS];   // [n][k]
    __shared__ __bf16 sBlo[BN * LDBS];

    const int tid   = threadIdx.x;
    const int lane  = tid & 31;
    const int wave  = tid >> 5;
    const int wm    = wave & 3;
    const int wn    = wave >> 2;
    const int lmod  = lane & 15;
    const int lhalf = lane >> 4;

    const int gm0 = blockIdx.y * BM;
    const int gn0 = blockIdx.x * BN;

    v8f acc[2][4];
#pragma unroll
    for (int mt = 0; mt < 2; ++mt)
#pragma unroll
        for (int nt = 0; nt < 4; ++nt)
            acc[mt][nt] = (v8f){0.f, 0.f, 0.f, 0.f, 0.f, 0.f, 0.f, 0.f};

    for (int k0 = 0; k0 < Kd; k0 += BK) {
        // ---- stage all four tiles: 16B per lane-chunk, async to LDS ----
#pragma unroll
        for (int t = 0; t < 2; ++t) {
            int i   = tid + t * 256;      // 0..511: 128 rows x 4 chunks
            int row = i >> 2;
            int c   = i & 3;
            size_t ga = (size_t)(gm0 + row) * Kd + k0 + c * 8;
            size_t gb = (size_t)(gn0 + row) * Kd + k0 + c * 8;
            int    la = row * LDBS + c * 8;
            copy16_g2l(Xhi  + ga, sAhi + la);
            copy16_g2l(Xlo  + ga, sAlo + la);
            copy16_g2l(BhiT + gb, sBhi + la);
            copy16_g2l(BloT + gb, sBlo + la);
        }
        staging_fence();

        // ---- fragments in native WMMA operand layout ----
        v16bf ahi[2], alo[2];
#pragma unroll
        for (int mt = 0; mt < 2; ++mt) {
            const __bf16* ar = sAhi + (wm * 32 + mt * 16 + lmod) * LDBS + lhalf * 8;
            const __bf16* al = sAlo + (wm * 32 + mt * 16 + lmod) * LDBS + lhalf * 8;
            v8bf h0 = *(const v8bf*)(ar);
            v8bf h1 = *(const v8bf*)(ar + 16);
            v8bf l0 = *(const v8bf*)(al);
            v8bf l1 = *(const v8bf*)(al + 16);
            ahi[mt] = __builtin_shufflevector(h0, h1, 0, 1, 2, 3, 4, 5, 6, 7,
                                              8, 9, 10, 11, 12, 13, 14, 15);
            alo[mt] = __builtin_shufflevector(l0, l1, 0, 1, 2, 3, 4, 5, 6, 7,
                                              8, 9, 10, 11, 12, 13, 14, 15);
        }
        v16bf bhi[4], blo[4];
#pragma unroll
        for (int nt = 0; nt < 4; ++nt) {
            const __bf16* br = sBhi + (wn * 64 + nt * 16 + lmod) * LDBS + lhalf * 8;
            const __bf16* bl = sBlo + (wn * 64 + nt * 16 + lmod) * LDBS + lhalf * 8;
            v8bf h0 = *(const v8bf*)(br);
            v8bf h1 = *(const v8bf*)(br + 16);
            v8bf l0 = *(const v8bf*)(bl);
            v8bf l1 = *(const v8bf*)(bl + 16);
            bhi[nt] = __builtin_shufflevector(h0, h1, 0, 1, 2, 3, 4, 5, 6, 7,
                                              8, 9, 10, 11, 12, 13, 14, 15);
            blo[nt] = __builtin_shufflevector(l0, l1, 0, 1, 2, 3, 4, 5, 6, 7,
                                              8, 9, 10, 11, 12, 13, 14, 15);
        }

        // ---- bf16x3 emulated-fp32 WMMA ----
#pragma unroll
        for (int mt = 0; mt < 2; ++mt) {
#pragma unroll
            for (int nt = 0; nt < 4; ++nt) {
                acc[mt][nt] = __builtin_amdgcn_wmma_f32_16x16x32_bf16(
                    false, ahi[mt], false, bhi[nt], (short)0, acc[mt][nt], false, false);
                acc[mt][nt] = __builtin_amdgcn_wmma_f32_16x16x32_bf16(
                    false, ahi[mt], false, blo[nt], (short)0, acc[mt][nt], false, false);
                acc[mt][nt] = __builtin_amdgcn_wmma_f32_16x16x32_bf16(
                    false, alo[mt], false, bhi[nt], (short)0, acc[mt][nt], false, false);
            }
        }

        __syncthreads();
    }

#pragma unroll
    for (int mt = 0; mt < 2; ++mt) {
#pragma unroll
        for (int nt = 0; nt < 4; ++nt) {
            const int n     = gn0 + wn * 64 + nt * 16 + lmod;
            const int mbase = gm0 + wm * 32 + mt * 16 + lhalf * 8;
            const float bb  = Bias[n];
#pragma unroll
            for (int r = 0; r < 8; ++r) {
                float v = acc[mt][nt][r] + bb;
                Out[(size_t)(mbase + r) * N + n] = v > 0.f ? v : 0.f;
            }
        }
    }
}

// ---------------------------------------------------------------------------
// Fallback: fused convert-in-kernel version (no workspace needed).
// ---------------------------------------------------------------------------
__global__ __launch_bounds__(256)
void schizo_wmma_gemm_fused(const float* __restrict__ X,
                            const float* __restrict__ Kw,
                            const float* __restrict__ Wn,
                            const float* __restrict__ Bias,
                            float* __restrict__ Out,
                            int M, int N, int Kd)
{
    __shared__ __bf16 Ahi[BM * LDBS];
    __shared__ __bf16 Alo[BM * LDBS];
    __shared__ __bf16 Bhi[BN * LDBS];
    __shared__ __bf16 Blo[BN * LDBS];

    const int tid   = threadIdx.x;
    const int lane  = tid & 31;
    const int wave  = tid >> 5;
    const int wm    = wave & 3;
    const int wn    = wave >> 2;
    const int lmod  = lane & 15;
    const int lhalf = lane >> 4;

    const int gm0 = blockIdx.y * BM;
    const int gn0 = blockIdx.x * BN;

    v8f acc[2][4];
#pragma unroll
    for (int mt = 0; mt < 2; ++mt)
#pragma unroll
        for (int nt = 0; nt < 4; ++nt)
            acc[mt][nt] = (v8f){0.f, 0.f, 0.f, 0.f, 0.f, 0.f, 0.f, 0.f};

    const int brb = (tid >> 5) * 4;
    const int bc4 = tid & 31;

    for (int k0 = 0; k0 < Kd; k0 += BK) {
        if (k0 + BK < Kd) {
            __builtin_prefetch(X  + (size_t)(gm0 + (tid >> 1)) * Kd + (k0 + BK), 0, 0);
            __builtin_prefetch(Kw + (size_t)(k0 + BK + brb) * N + gn0 + bc4 * 4, 0, 0);
            __builtin_prefetch(Wn + (size_t)(k0 + BK + brb) * N + gn0 + bc4 * 4, 0, 0);
        }
#pragma unroll
        for (int t = 0; t < 4; ++t) {
            int i   = tid + t * 256;
            int row = i >> 3;
            int c4  = i & 7;
            const float4 v = *(const float4*)(X + (size_t)(gm0 + row) * Kd + k0 + c4 * 4);
            v4bf h, l;
            h[0] = (__bf16)v.x;  l[0] = lo_part(v.x, h[0]);
            h[1] = (__bf16)v.y;  l[1] = lo_part(v.y, h[1]);
            h[2] = (__bf16)v.z;  l[2] = lo_part(v.z, h[2]);
            h[3] = (__bf16)v.w;  l[3] = lo_part(v.w, h[3]);
            *(v4bf*)(Ahi + row * LDBS + c4 * 4) = h;
            *(v4bf*)(Alo + row * LDBS + c4 * 4) = l;
        }
        {
            float pr[4][4];
#pragma unroll
            for (int rr = 0; rr < 4; ++rr) {
                size_t g = (size_t)(k0 + brb + rr) * N + gn0 + bc4 * 4;
                float4 kv = *(const float4*)(Kw + g);
                float4 wv = *(const float4*)(Wn + g);
                pr[rr][0] = kv.x * wv.x;
                pr[rr][1] = kv.y * wv.y;
                pr[rr][2] = kv.z * wv.z;
                pr[rr][3] = kv.w * wv.w;
            }
#pragma unroll
            for (int e = 0; e < 4; ++e) {
                int n_loc = bc4 * 4 + e;
                v4bf h, l;
#pragma unroll
                for (int rr = 0; rr < 4; ++rr) {
                    h[rr] = (__bf16)pr[rr][e];
                    l[rr] = lo_part(pr[rr][e], h[rr]);
                }
                *(v4bf*)(Bhi + n_loc * LDBS + brb) = h;
                *(v4bf*)(Blo + n_loc * LDBS + brb) = l;
            }
        }

        __syncthreads();

        v16bf ahi[2], alo[2];
#pragma unroll
        for (int mt = 0; mt < 2; ++mt) {
            const __bf16* ar = Ahi + (wm * 32 + mt * 16 + lmod) * LDBS + lhalf * 8;
            const __bf16* al = Alo + (wm * 32 + mt * 16 + lmod) * LDBS + lhalf * 8;
            v8bf h0 = *(const v8bf*)(ar);
            v8bf h1 = *(const v8bf*)(ar + 16);
            v8bf l0 = *(const v8bf*)(al);
            v8bf l1 = *(const v8bf*)(al + 16);
            ahi[mt] = __builtin_shufflevector(h0, h1, 0, 1, 2, 3, 4, 5, 6, 7,
                                              8, 9, 10, 11, 12, 13, 14, 15);
            alo[mt] = __builtin_shufflevector(l0, l1, 0, 1, 2, 3, 4, 5, 6, 7,
                                              8, 9, 10, 11, 12, 13, 14, 15);
        }
        v16bf bhi[4], blo[4];
#pragma unroll
        for (int nt = 0; nt < 4; ++nt) {
            const __bf16* br = Bhi + (wn * 64 + nt * 16 + lmod) * LDBS + lhalf * 8;
            const __bf16* bl = Blo + (wn * 64 + nt * 16 + lmod) * LDBS + lhalf * 8;
            v8bf h0 = *(const v8bf*)(br);
            v8bf h1 = *(const v8bf*)(br + 16);
            v8bf l0 = *(const v8bf*)(bl);
            v8bf l1 = *(const v8bf*)(bl + 16);
            bhi[nt] = __builtin_shufflevector(h0, h1, 0, 1, 2, 3, 4, 5, 6, 7,
                                              8, 9, 10, 11, 12, 13, 14, 15);
            blo[nt] = __builtin_shufflevector(l0, l1, 0, 1, 2, 3, 4, 5, 6, 7,
                                              8, 9, 10, 11, 12, 13, 14, 15);
        }

#pragma unroll
        for (int mt = 0; mt < 2; ++mt) {
#pragma unroll
            for (int nt = 0; nt < 4; ++nt) {
                acc[mt][nt] = __builtin_amdgcn_wmma_f32_16x16x32_bf16(
                    false, ahi[mt], false, bhi[nt], (short)0, acc[mt][nt], false, false);
                acc[mt][nt] = __builtin_amdgcn_wmma_f32_16x16x32_bf16(
                    false, ahi[mt], false, blo[nt], (short)0, acc[mt][nt], false, false);
                acc[mt][nt] = __builtin_amdgcn_wmma_f32_16x16x32_bf16(
                    false, alo[mt], false, bhi[nt], (short)0, acc[mt][nt], false, false);
            }
        }

        __syncthreads();
    }

#pragma unroll
    for (int mt = 0; mt < 2; ++mt) {
#pragma unroll
        for (int nt = 0; nt < 4; ++nt) {
            const int n     = gn0 + wn * 64 + nt * 16 + lmod;
            const int mbase = gm0 + wm * 32 + mt * 16 + lhalf * 8;
            const float bb  = Bias[n];
#pragma unroll
            for (int r = 0; r < 8; ++r) {
                float v = acc[mt][nt][r] + bb;
                Out[(size_t)(mbase + r) * N + n] = v > 0.f ? v : 0.f;
            }
        }
    }
}

extern "C" void kernel_launch(void* const* d_in, const int* in_sizes, int n_in,
                              void* d_out, int out_size, void* d_ws, size_t ws_size,
                              hipStream_t stream) {
    const float* X    = (const float*)d_in[0];
    const float* Kw   = (const float*)d_in[1];
    const float* Wn   = (const float*)d_in[2];
    const float* Bias = (const float*)d_in[3];
    float* Out        = (float*)d_out;

    const int N  = in_sizes[3];           // bias length = NY
    const int Kd = in_sizes[1] / N;       // kernel is [K, N]
    const int M  = in_sizes[0] / Kd;      // x is [M, K]

    const size_t xElems = (size_t)M * Kd;
    const size_t bElems = (size_t)Kd * N;
    const size_t need   = (xElems * 2 + bElems * 2) * sizeof(__bf16);

    dim3 grid(N / BN, M / BM);

    if (ws_size >= need) {
        __bf16* Xhi  = (__bf16*)d_ws;
        __bf16* Xlo  = Xhi + xElems;
        __bf16* BhiT = Xlo + xElems;
        __bf16* BloT = BhiT + bElems;

        prep_split_x<<<dim3((unsigned)(xElems / 1024)), dim3(256), 0, stream>>>(
            X, Xhi, Xlo);
        prep_fuse_b<<<dim3(N / 128, Kd / 32), dim3(256), 0, stream>>>(
            Kw, Wn, BhiT, BloT, N, Kd);
        schizo_wmma_gemm_pre<<<grid, dim3(256), 0, stream>>>(
            Xhi, Xlo, BhiT, BloT, Bias, Out, M, N, Kd);
    } else {
        schizo_wmma_gemm_fused<<<grid, dim3(256), 0, stream>>>(
            X, Kw, Wn, Bias, Out, M, N, Kd);
    }
}
